// ContrastiveLoss_78185584657073
// MI455X (gfx1250) — matrix operand order repaired and checked by reading, not compiled, not run on hardware
//
#include <hip/hip_runtime.h>

typedef _Float16 v8h  __attribute__((ext_vector_type(8)));
typedef _Float16 v16h __attribute__((ext_vector_type(16)));
typedef float    v8f  __attribute__((ext_vector_type(8)));

#define NROWS 8192
#define DDIM  256
#define NTILE (NROWS / 16)      // 512 16x16 tiles per dimension
#define NPAIR (NROWS / 32)      // 256 32-row I-blocks in total
#define JCHUNK 32               // column tiles handled per wave
#define NJC   (NTILE / JCHUNK)  // 16 J-chunks
#define MARGIN 1.0f

union H16 { v8h p[2]; v16h v; };

// ---------------- zero the accumulators ----------------
__global__ void CL_init(float* acc) {
    if (threadIdx.x < 3) acc[threadIdx.x] = 0.0f;
}

// ---------------- L2-normalize rows, f32 -> f16 ----------------
// one wave32 per row; each lane handles 8 contiguous columns
__global__ void __launch_bounds__(256)
CL_normalize(const float* __restrict__ emb, _Float16* __restrict__ out) {
    const int wave = (blockIdx.x * blockDim.x + threadIdx.x) >> 5;
    const int lane = threadIdx.x & 31;
    if (wave >= NROWS) return;

    const float4* row = (const float4*)(emb + (size_t)wave * DDIM);
    float4 x0 = row[lane * 2 + 0];
    float4 x1 = row[lane * 2 + 1];

    float s = x0.x * x0.x + x0.y * x0.y + x0.z * x0.z + x0.w * x0.w
            + x1.x * x1.x + x1.y * x1.y + x1.z * x1.z + x1.w * x1.w;
    #pragma unroll
    for (int off = 16; off > 0; off >>= 1)
        s += __shfl_xor(s, off, 32);

    const float inv = rsqrtf(s);
    v8h h;
    h[0] = (_Float16)(x0.x * inv); h[1] = (_Float16)(x0.y * inv);
    h[2] = (_Float16)(x0.z * inv); h[3] = (_Float16)(x0.w * inv);
    h[4] = (_Float16)(x1.x * inv); h[5] = (_Float16)(x1.y * inv);
    h[6] = (_Float16)(x1.z * inv); h[7] = (_Float16)(x1.w * inv);
    *(v8h*)(out + (size_t)wave * DDIM + lane * 8) = h;
}

// ---------------- fused E*E^T + masked loss reduction ----------------
// wave w: 32-row I-block it = w>>4 (i0 = 32*it), column tiles jc*32 .. +31.
// Two A row-tiles (full K=256, 128 VGPRs) held in registers; every B
// fragment feeds two WMMAs -> 32 FLOP per L2 byte.
__global__ void __launch_bounds__(256)
CL_sim_loss(const _Float16* __restrict__ e, const int* __restrict__ labels,
            float* __restrict__ acc) {
    const int wave = (blockIdx.x << 3) + (threadIdx.x >> 5);
    const int lane = threadIdx.x & 31;
    const int it   = wave >> 4;         // 0 .. 255
    const int jc   = wave & (NJC - 1);  // 0 .. 15
    const int i0   = it * 32;
    const int half = lane >> 4;         // 0 or 1
    const int l15  = lane & 15;

    // --- load A fragments: ISA 16-bit A 16x32 layout ---
    // lane (half,l15): row i0+l15, halfs at K = kc*32 + half*8 + [0..7]
    //                               and K = kc*32 + 16 + half*8 + [0..7]
    H16 A0[8], A1[8];
    const _Float16* arow0 = e + (size_t)(i0 + l15) * DDIM + half * 8;
    const _Float16* arow1 = arow0 + (size_t)16 * DDIM;
    #pragma unroll
    for (int kc = 0; kc < 8; ++kc) {
        A0[kc].p[0] = *(const v8h*)(arow0 + kc * 32);
        A0[kc].p[1] = *(const v8h*)(arow0 + kc * 32 + 16);
        A1[kc].p[0] = *(const v8h*)(arow1 + kc * 32);
        A1[kc].p[1] = *(const v8h*)(arow1 + kc * 32 + 16);
    }

    // row labels for the epilogue: C-layout row m = half*8 + r
    int lab0[8], lab1[8];
    #pragma unroll
    for (int r = 0; r < 8; ++r) {
        lab0[r] = labels[i0 + half * 8 + r];
        lab1[r] = labels[i0 + 16 + half * 8 + r];
    }

    float pos_sum = 0.0f, neg_sum = 0.0f, pos_cnt = 0.0f;

    for (int jt = 0; jt < JCHUNK; ++jt) {
        const int j0 = (jc * JCHUNK + jt) * 16;
        // B 32x16 layout: lane (half,l15) holds column l15,
        // K = kc*32 + half*8 + [0..7] and kc*32 + 16 + half*8 + [0..7]
        const _Float16* brow = e + (size_t)(j0 + l15) * DDIM + half * 8;

        v8f c0 = {}, c1 = {};
        #pragma unroll
        for (int kc = 0; kc < 8; ++kc) {
            H16 B;
            B.p[0] = *(const v8h*)(brow + kc * 32);
            B.p[1] = *(const v8h*)(brow + kc * 32 + 16);
            c0 = __builtin_amdgcn_wmma_f32_16x16x32_f16(
                     false, A0[kc].v, false, B.v, (short)0, c0, false, false);
            c1 = __builtin_amdgcn_wmma_f32_16x16x32_f16(
                     false, A1[kc].v, false, B.v, (short)0, c1, false, false);
        }

        const int lab_j = labels[j0 + l15];
        #pragma unroll
        for (int r = 0; r < 8; ++r) {
            {
                const float s   = c0[r];
                const bool  pos = (lab0[r] == lab_j);
                const float d1  = 1.0f - s;
                const float d2  = fmaxf(s - MARGIN, 0.0f);
                pos_sum += pos ? d1 * d1 : 0.0f;
                neg_sum += pos ? 0.0f : d2 * d2;
                pos_cnt += pos ? 1.0f : 0.0f;
            }
            {
                const float s   = c1[r];
                const bool  pos = (lab1[r] == lab_j);
                const float d1  = 1.0f - s;
                const float d2  = fmaxf(s - MARGIN, 0.0f);
                pos_sum += pos ? d1 * d1 : 0.0f;
                neg_sum += pos ? 0.0f : d2 * d2;
                pos_cnt += pos ? 1.0f : 0.0f;
            }
        }
    }

    // wave32 reduction then one atomic per wave
    #pragma unroll
    for (int off = 16; off > 0; off >>= 1) {
        pos_sum += __shfl_xor(pos_sum, off, 32);
        neg_sum += __shfl_xor(neg_sum, off, 32);
        pos_cnt += __shfl_xor(pos_cnt, off, 32);
    }
    if (lane == 0) {
        atomicAdd(&acc[0], pos_sum);
        atomicAdd(&acc[1], neg_sum);
        atomicAdd(&acc[2], pos_cnt);
    }
}

// ---------------- final scalar ----------------
__global__ void CL_finalize(const float* __restrict__ acc, float* __restrict__ out) {
    const float pos_cnt = acc[2];
    const float neg_cnt = (float)NROWS * (float)NROWS - pos_cnt;
    out[0] = acc[0] / pos_cnt + acc[1] / neg_cnt;
}

extern "C" void kernel_launch(void* const* d_in, const int* in_sizes, int n_in,
                              void* d_out, int out_size, void* d_ws, size_t ws_size,
                              hipStream_t stream) {
    const float* emb    = (const float*)d_in[0];
    const int*   labels = (const int*)d_in[1];
    float*       out    = (float*)d_out;

    float*     acc   = (float*)d_ws;                       // 3 accumulators
    _Float16*  enorm = (_Float16*)((char*)d_ws + 256);     // 8192*256 f16 = 4 MB

    CL_init<<<1, 32, 0, stream>>>(acc);
    CL_normalize<<<NROWS / 8, 256, 0, stream>>>(emb, enorm);
    // 256 I-blocks * 16 J-chunks = 4096 waves, 8 waves per 256-thread block
    CL_sim_loss<<<(NPAIR * NJC) / 8, 256, 0, stream>>>(enorm, labels, acc);
    CL_finalize<<<1, 1, 0, stream>>>(acc, out);
}
